// Head_32959579029970
// MI455X (gfx1250) — matrix operand order repaired and checked by reading, not compiled, not run on hardware
//
#include <hip/hip_runtime.h>

typedef __attribute__((ext_vector_type(16))) __bf16        v16bf;
typedef __attribute__((ext_vector_type(8)))  float         v8f;
typedef __attribute__((ext_vector_type(4)))  int           v4i;

#define B_DIM 4
#define T_DIM 4096
#define C_DIM 1024
#define H_DIM 128

// LDS strides in ushort elements. All strides are multiples of 8 ushorts so
// every row starts 16B-aligned (b128-friendly), and the byte stride is an odd
// multiple of 16B so row-indexed accesses sweep banks without conflicts.
constexpr int XS_STRIDE = 40;   // x tile rows    (80B  = 5*16)
constexpr int WT_STRIDE = 40;   // W^T tile rows  (80B)
constexpr int KS_STRIDE = 136;  // K/Q tile rows  (272B = 17*16)
constexpr int VT_STRIDE = 72;   // V^T tile rows  (144B = 9*16)
constexpr int PS_STRIDE = 72;   // P tile rows    (144B)
constexpr int TS_STRIDE = 72;   // proj transpose tile rows (144B)

struct U4x2 { uint4 a, b; };    // 32B container for one v16bf fragment

// typed pointers for the async-to-LDS builtin:
// param0 = global (AS1) int4*, param1 = LDS (AS3) int4*
typedef __attribute__((address_space(1))) v4i v4i_g;
typedef __attribute__((address_space(3))) v4i v4i_l;

#if __has_builtin(__builtin_amdgcn_global_load_async_to_lds_b128) && \
    __has_builtin(__builtin_amdgcn_s_wait_asynccnt)
#define HAVE_ASYNC 1
#else
#define HAVE_ASYNC 0
#endif

// ---- fast float->bf16 (RNE) via V_FMA_MIXLO/MIXHI_BF16 (VOP3P op 62/63) ----
__device__ __forceinline__ unsigned int pack2(float lo, float hi) {
    unsigned int d;
    asm("v_fma_mixlo_bf16 %0, %1, 1.0, 0" : "=v"(d) : "v"(lo));
    asm("v_fma_mixhi_bf16 %0, %1, 1.0, 0" : "+v"(d) : "v"(hi));
    return d;
}
__device__ __forceinline__ unsigned short f2bf(float f) {
    unsigned int d;
    asm("v_fma_mixlo_bf16 %0, %1, 1.0, 0" : "=v"(d) : "v"(f));
    return (unsigned short)d;
}

// ---- fast base-2 exponential (hardware transcendental) ---------------------
#if __has_builtin(__builtin_amdgcn_exp2f)
__device__ __forceinline__ float fexp2(float x) { return __builtin_amdgcn_exp2f(x); }
#else
__device__ __forceinline__ float fexp2(float x) { return exp2f(x); }
#endif

// A-matrix 16x32 bf16 fragment. Per lane the data is one contiguous 32B
// run: K = kbase+lh*16+{0..15}.
__device__ __forceinline__ v16bf load_afrag(const unsigned short* rowbase,
                                            int kbase, int lh) {
    const unsigned short* p = rowbase + kbase + lh * 16;
    U4x2 u;
    u.a = *(const uint4*)p;
    u.b = *(const uint4*)(p + 8);
    return __builtin_bit_cast(v16bf, u);
}

// B-matrix 32x16 bf16 fragment: this lane's 16 contraction values are
// contiguous in an LDS row (32B): p -> K = k0+{0..15} at fixed column.
__device__ __forceinline__ v16bf load_bfrag(const unsigned short* p) {
    U4x2 u;
    u.a = *(const uint4*)p;
    u.b = *(const uint4*)(p + 8);
    return __builtin_bit_cast(v16bf, u);
}

// ---- dense tile copies, per-wave work split (for async prefetch) -----------
// K tile: 64 rows x 256B  -> Ks (stride KS_STRIDE)
// V tile: 128 rows x 128B -> Vt (stride VT_STRIDE), source row stride 4096
#if HAVE_ASYNC
__device__ __forceinline__ void prefetch_k(unsigned short* dst,
                                           const unsigned short* src,
                                           int wv, int lane) {
#pragma unroll
    for (int i = 0; i < 8; ++i) {
        int chunk = (wv * 8 + i) * 32 + lane;      // 0..1023
        int r = chunk >> 4, c = chunk & 15;
        __builtin_amdgcn_global_load_async_to_lds_b128(
            (v4i_g*)(void*)const_cast<unsigned short*>(src + r * 128 + c * 8),
            (v4i_l*)(void*)(dst + r * KS_STRIDE + c * 8), 0, 0);
    }
}
__device__ __forceinline__ void prefetch_v(unsigned short* dst,
                                           const unsigned short* src,
                                           int wv, int lane) {
#pragma unroll
    for (int i = 0; i < 8; ++i) {
        int chunk = (wv * 8 + i) * 32 + lane;      // 0..1023
        int r = chunk >> 3, c = chunk & 7;
        __builtin_amdgcn_global_load_async_to_lds_b128(
            (v4i_g*)(void*)const_cast<unsigned short*>(src + (size_t)r * T_DIM + c * 8),
            (v4i_l*)(void*)(dst + r * VT_STRIDE + c * 8), 0, 0);
    }
}
#else
__device__ __forceinline__ void copy_k(unsigned short* dst,
                                       const unsigned short* src, int t) {
#pragma unroll
    for (int i = 0; i < 8; ++i) {
        int chunk = i * 128 + t;
        int r = chunk >> 4, c = chunk & 15;
        *(uint4*)(dst + r * KS_STRIDE + c * 8) =
            *(const uint4*)(src + r * 128 + c * 8);
    }
}
__device__ __forceinline__ void copy_v(unsigned short* dst,
                                       const unsigned short* src, int t) {
#pragma unroll
    for (int i = 0; i < 8; ++i) {
        int chunk = i * 128 + t;
        int r = chunk >> 3, c = chunk & 7;
        *(uint4*)(dst + r * VT_STRIDE + c * 8) =
            *(const uint4*)(src + (size_t)r * T_DIM + c * 8);
    }
}
#endif

// Q tile copy (dense, row-major, synchronous -- done once per block).
__device__ __forceinline__ void copy_q(unsigned short* dst,
                                       const unsigned short* src, int t) {
#pragma unroll
    for (int i = 0; i < 8; ++i) {
        int chunk = i * 128 + t;
        int r = chunk >> 4, c = chunk & 15;
        *(uint4*)(dst + r * KS_STRIDE + c * 8) =
            *(const uint4*)(src + r * 128 + c * 8);
    }
}

// ---------------------------------------------------------------------------
// Projection: y = scale * (x @ W), y is bf16.
// tmode==0: write y[row][h]   (row-major, for Q and K)
// tmode==1: write yT[b][h][t] (transposed per batch, for V -- makes the
//           attention-side V staging a dense copy in B-fragment layout)
// Block: 64 rows x 128 cols, 4 wave32 waves (16 rows each). K-step = 32.
// ---------------------------------------------------------------------------
__global__ void __launch_bounds__(128) proj_kernel(const float* __restrict__ x,
                                                   const float* __restrict__ W,
                                                   unsigned short* __restrict__ out,
                                                   float scale, int tmode) {
    __shared__ unsigned short xs[64 * XS_STRIDE];
    __shared__ unsigned short wt[128 * WT_STRIDE];   // W^T tile, c-pairs packed
    __shared__ unsigned short ts[128 * TS_STRIDE];   // transpose staging (tmode)

    const int t    = threadIdx.x;
    const int lane = t & 31;
    const int wv   = t >> 5;
    const int lh   = lane >> 4;
    const int lr   = lane & 15;
    const int rowbase = blockIdx.x * 64;

    v8f acc[8] = {};

    for (int kt = 0; kt < C_DIM / 32; ++kt) {
        // stage x tile (64 x 32 f32 -> bf16, row-major)
        {
            int r  = t >> 1;
            int c0 = (t & 1) * 16;
            const float* src = x + (size_t)(rowbase + r) * C_DIM + kt * 32 + c0;
            unsigned int* d = (unsigned int*)&xs[r * XS_STRIDE + c0];
#pragma unroll
            for (int i = 0; i < 4; ++i) {
                float4 f = ((const float4*)src)[i];
                d[2 * i]     = pack2(f.x, f.y);
                d[2 * i + 1] = pack2(f.z, f.w);
            }
        }
        // stage W tile (32 x 128 f32) transposed -> wt[h][c], c-pairs packed
        {
            int rp = t & 15;          // c pair -> c = 2rp, 2rp+1
            int h0 = (t >> 4) * 16;   // 16 h-columns
            const float* s0 = W + (size_t)(kt * 32 + 2 * rp) * H_DIM + h0;
            const float* s1 = s0 + H_DIM;
#pragma unroll
            for (int i = 0; i < 16; ++i)
                *(unsigned int*)&wt[(h0 + i) * WT_STRIDE + 2 * rp] =
                    pack2(s0[i], s1[i]);
        }
        __syncthreads();

        v16bf a = load_afrag(&xs[(wv * 16 + lr) * XS_STRIDE], 0, lh);
#pragma unroll
        for (int nt = 0; nt < 8; ++nt) {
            v16bf b = load_bfrag(&wt[(nt * 16 + lr) * WT_STRIDE + lh * 16]);
            acc[nt] = __builtin_amdgcn_wmma_f32_16x16x32_bf16(
                false, a, false, b, (short)0, acc[nt], false, false);
        }
        __syncthreads();
    }

    if (tmode == 0) {
#pragma unroll
        for (int nt = 0; nt < 8; ++nt)
#pragma unroll
            for (int i = 0; i < 8; ++i) {
                int row = rowbase + wv * 16 + i + lh * 8;
                int col = nt * 16 + lr;
                out[(size_t)row * H_DIM + col] = f2bf(acc[nt][i] * scale);
            }
    } else {
        // transpose through LDS, then coalesced b128 row stores to yT[b][h][t]
#pragma unroll
        for (int nt = 0; nt < 8; ++nt)
#pragma unroll
            for (int i = 0; i < 8; ++i)
                ts[(nt * 16 + lr) * TS_STRIDE + wv * 16 + i + lh * 8] =
                    f2bf(acc[nt][i] * scale);
        __syncthreads();
        const int b    = rowbase >> 12;       // / T_DIM
        const int tloc = rowbase & (T_DIM - 1);
        unsigned short* drow = out + ((size_t)(b * H_DIM + t)) * T_DIM + tloc;
#pragma unroll
        for (int c = 0; c < 8; ++c)
            *(uint4*)(drow + c * 8) = *(const uint4*)(&ts[t * TS_STRIDE + c * 8]);
    }
}

// ---------------------------------------------------------------------------
// Flash attention over bf16 q/k (row-major) and vT (transposed), causal,
// f32 output. q pre-scaled by log2(e)/sqrt(H) (base-2 softmax domain).
// Block: 64 query rows (4 waves x 16). Double-buffered LDS tiles with
// GLOBAL_LOAD_ASYNC_TO_LDS_B128 prefetch of the next key tile (ASYNCcnt).
// ---------------------------------------------------------------------------
__global__ void __launch_bounds__(128) attn_kernel(const unsigned short* __restrict__ qg,
                                                   const unsigned short* __restrict__ kg,
                                                   const unsigned short* __restrict__ vtg,
                                                   float* __restrict__ out) {
    __shared__ unsigned short KsBuf[2][64 * KS_STRIDE];
    __shared__ unsigned short VtBuf[2][128 * VT_STRIDE];
    __shared__ unsigned short Ps[4][16 * PS_STRIDE];

    const int t    = threadIdx.x;
    const int lane = t & 31;
    const int wv   = t >> 5;
    const int lh   = lane >> 4;
    const int lr   = lane & 15;
    const int qi   = blockIdx.x;                          // query tile (64 rows)
    const size_t base  = (size_t)blockIdx.y * T_DIM * H_DIM;   // q/k/out batch
    const unsigned short* vtb = vtg + (size_t)blockIdx.y * H_DIM * T_DIM;

#if HAVE_ASYNC
    // kick off async prefetch of key tile 0 while we process Q
    prefetch_k(KsBuf[0], kg + base, wv, lane);
    prefetch_v(VtBuf[0], vtb, wv, lane);
#endif

    // Stage Q tile through KsBuf[1], build 8 A-fragments (8 K-steps of 32)
    copy_q(KsBuf[1], qg + base + (size_t)qi * 64 * H_DIM, t);
    __syncthreads();
    v16bf a_q[8];
#pragma unroll
    for (int ks = 0; ks < 8; ++ks)
        a_q[ks] = load_afrag(&KsBuf[1][(wv * 16 + lr) * KS_STRIDE], ks * 32, lh);
    __syncthreads();

    v8f   o[8] = {};
    float mprev[8], lsum[8];
#pragma unroll
    for (int i = 0; i < 8; ++i) { mprev[i] = -__builtin_inff(); lsum[i] = 0.f; }

    for (int j = 0; j <= qi; ++j) {
        const unsigned short* ks = KsBuf[j & 1];
        const unsigned short* vt = VtBuf[j & 1];

#if HAVE_ASYNC
        if (j < qi) {   // prefetch next tile into the other buffer
            prefetch_k(KsBuf[(j + 1) & 1], kg + base + (size_t)(j + 1) * 64 * H_DIM,
                       wv, lane);
            prefetch_v(VtBuf[(j + 1) & 1], vtb + (j + 1) * 64, wv, lane);
            __builtin_amdgcn_s_wait_asynccnt(16);  // current tile's 16 retired
        } else {
            __builtin_amdgcn_s_wait_asynccnt(0);
        }
        __syncthreads();
#else
        copy_k(KsBuf[j & 1], kg + base + (size_t)j * 64 * H_DIM, t);
        copy_v(VtBuf[j & 1], vtb + j * 64, t);
        __syncthreads();
#endif

        // S = Q K^T  (16 x 64 per wave)
        v8f s[4] = {};
#pragma unroll
        for (int nt = 0; nt < 4; ++nt)
#pragma unroll
            for (int kk = 0; kk < 8; ++kk) {
                v16bf b = load_bfrag(
                    &ks[(nt * 16 + lr) * KS_STRIDE + kk * 32 + lh * 16]);
                s[nt] = __builtin_amdgcn_wmma_f32_16x16x32_bf16(
                    false, a_q[kk], false, b, (short)0, s[nt], false, false);
            }

        // causal mask on the diagonal tile
        if (j == qi) {
#pragma unroll
            for (int nt = 0; nt < 4; ++nt)
#pragma unroll
                for (int i = 0; i < 8; ++i) {
                    int qrow = wv * 16 + i + lh * 8;
                    int krow = nt * 16 + lr;
                    if (krow > qrow) s[nt][i] = -__builtin_inff();
                }
        }

        // online softmax, log2 domain
#pragma unroll
        for (int i = 0; i < 8; ++i) {
            float cm = fmaxf(fmaxf(s[0][i], s[1][i]), fmaxf(s[2][i], s[3][i]));
#pragma unroll
            for (int m = 1; m < 16; m <<= 1) cm = fmaxf(cm, __shfl_xor(cm, m, 32));
            float mnew  = fmaxf(mprev[i], cm);
            float alpha = fexp2(mprev[i] - mnew);
            float psum  = 0.f;
#pragma unroll
            for (int nt = 0; nt < 4; ++nt) {
                float p = fexp2(s[nt][i] - mnew);
                s[nt][i] = p;
                psum += p;
            }
#pragma unroll
            for (int m = 1; m < 16; m <<= 1) psum += __shfl_xor(psum, m, 32);
            lsum[i]  = lsum[i] * alpha + psum;
            mprev[i] = mnew;
#pragma unroll
            for (int nt2 = 0; nt2 < 8; ++nt2) o[nt2][i] *= alpha;
        }

        // P (C-layout) -> per-wave LDS -> A-fragments (wave-private)
        unsigned short* pw = Ps[wv];
#pragma unroll
        for (int nt = 0; nt < 4; ++nt)
#pragma unroll
            for (int i = 0; i < 8; ++i)
                pw[(i + 8 * lh) * PS_STRIDE + nt * 16 + lr] = f2bf(s[nt][i]);

        v16bf pf[2];
#pragma unroll
        for (int ks2 = 0; ks2 < 2; ++ks2)
            pf[ks2] = load_afrag(&pw[lr * PS_STRIDE], ks2 * 32, lh);

        // O += P V   (contraction over 64 keys -> 2 K-steps of 32)
#pragma unroll
        for (int nt2 = 0; nt2 < 8; ++nt2)
#pragma unroll
            for (int ks2 = 0; ks2 < 2; ++ks2) {
                v16bf bv = load_bfrag(
                    &vt[(nt2 * 16 + lr) * VT_STRIDE + ks2 * 32 + lh * 16]);
                o[nt2] = __builtin_amdgcn_wmma_f32_16x16x32_bf16(
                    false, pf[ks2], false, bv, (short)0, o[nt2], false, false);
            }
        __syncthreads();
    }

    // normalize and store f32 output
#pragma unroll
    for (int i = 0; i < 8; ++i) {
        float inv = 1.f / lsum[i];
#pragma unroll
        for (int nt2 = 0; nt2 < 8; ++nt2) {
            int row = qi * 64 + wv * 16 + i + 8 * lh;
            out[base + (size_t)row * H_DIM + nt2 * 16 + lr] = o[nt2][i] * inv;
        }
    }
}

extern "C" void kernel_launch(void* const* d_in, const int* in_sizes, int n_in,
                              void* d_out, int out_size, void* d_ws, size_t ws_size,
                              hipStream_t stream) {
    const float* x  = (const float*)d_in[0];
    const float* Wq = (const float*)d_in[1];
    const float* Wk = (const float*)d_in[2];
    const float* Wv = (const float*)d_in[3];
    float* out = (float*)d_out;

    const size_t qkv_elems = (size_t)B_DIM * T_DIM * H_DIM;
    unsigned short* qb  = (unsigned short*)d_ws;
    unsigned short* kb  = qb + qkv_elems;
    unsigned short* vbt = kb + qkv_elems;      // V stored transposed [b][h][t]

    // q scale = log2(e) / sqrt(H): base-2 softmax domain for raw v_exp_f32.
    const float scale_q = 1.4426950408889634f * 0.08838834764831845f;

    dim3 gp((B_DIM * T_DIM) / 64);             // 256 blocks
    proj_kernel<<<gp, 128, 0, stream>>>(x, Wq, qb,  scale_q, 0);
    proj_kernel<<<gp, 128, 0, stream>>>(x, Wk, kb,  1.0f,    0);
    proj_kernel<<<gp, 128, 0, stream>>>(x, Wv, vbt, 1.0f,    1);

    dim3 ga(T_DIM / 64, B_DIM);                // 64 x 4 blocks
    attn_kernel<<<ga, 128, 0, stream>>>(qb, kb, vbt, out);
}
